// VQVAE_86406152061159
// MI455X (gfx1250) — matrix operand order repaired and checked by reading, not compiled, not run on hardware
//
#include <hip/hip_runtime.h>
#include <hip/hip_bf16.h>

typedef __attribute__((ext_vector_type(16))) _Float16 v16h;
typedef __attribute__((ext_vector_type(8)))  float    v8f;

#define BN_EPS 1e-5f

// ---------------- workspace layout ----------------
// f16 (half-offset) region: NHWC activations, chunked weights, f16 codebook
static const size_t H_X4  = 0;         // x padded 3->4ch:  128*128*128*4 = 8388608
static const size_t H_A1  = 8388608;   // e1 out NHWC16:    128*64*64*16  = 8388608
static const size_t H_A2  = 16777216;  // e2 out NHWC16:    128*32*32*16  = 2097152
static const size_t H_A3  = 18874368;  // e3 out NHWC16:    128*17*17*16  = 591872
static const size_t H_ZE  = 19466240;  // e4 out NHWC32:    128*17*17*32  = 1183744
static const size_t H_ZQ  = 20649984;  // z_q NHWC32:       1183744
static const size_t H_B1  = 21833728;  // d1 out NHWC16:    591872
static const size_t H_B2  = 22425600;  // d2 out NHWC16:    128*34*34*16  = 2367488
static const size_t H_B3  = 24793088;  // d3 out NHWC16:    128*66*66*16  = 8921088
static const size_t H_B4  = 33714176;  // d4 out NHWC16:    128*132*132*16= 35684352
static const size_t H_WR  = 69398528;  // 8 layers * 4096 chunked-weight halfs
static const size_t H_EMB = 69431296;  // codebook f16: 512*32 = 16384
// f32 (float-offset) region starts at half 69447680 -> float 34723840
static const size_t F_STATS = 34723840; // 8 layers * 128 (sum[64] | sumsq[64])
static const size_t F_CNORM = 34724864; // 512   (total ws ~139 MB)

// ---------------- d_out layout (float offsets) ----------------
static const size_t OUT_XT  = 0;         // 128*3*132*132 = 6690816
static const size_t OUT_ZE  = 6690816;   // 128*32*17*17  = 1183744
static const size_t OUT_ZQ  = 7874560;   // 1183744
static const size_t OUT_LAT = 9058304;   // 36992

__global__ void zero_stats(float* st) {
  int i = blockIdx.x * 256 + threadIdx.x;
  if (i < 1024) st[i] = 0.f;
}

// =============== prep: x (NCHW f32, C=3) -> NHWC f16 padded to 4ch ===============
__global__ void prep_x4(const float* __restrict__ x, _Float16* __restrict__ out)
{
  int i = blockIdx.x * 256 + threadIdx.x;
  if (i >= 8388608) return;
  int c = i & 3;
  int pix = i >> 2;
  int w = pix % 128, h = (pix / 128) % 128, n = pix / 16384;
  float v = (c < 3) ? x[(((size_t)n * 3 + c) * 128 + h) * 128 + w] : 0.f;
  out[i] = (_Float16)v;
}

// =============== prep: weights -> per-WMMA-chunk f16 layout [chunk][co][local_k] ===============
// q = chunk*32 + local_k ; tap = q / CI ; ci = q % CI ; zero-pad invalid taps / channels.
__global__ void prep_wchunks(const float* __restrict__ w, _Float16* __restrict__ out,
                             int CIN_real, int CI, int COUT, int Kk, int transp, int total)
{
  int idx = blockIdx.x * 256 + threadIdx.x;
  if (idx >= total) return;
  int t  = idx / (COUT * 32);
  int r  = idx % (COUT * 32);
  int co = r / 32, lk = r % 32;
  int q = t * 32 + lk;
  int tap = q / CI, ci = q % CI;
  float v = 0.f;
  if (tap < Kk * Kk && ci < CIN_real) {
    int kh = tap / Kk, kw = tap % Kk;
    v = transp ? w[((ci * COUT + co) * Kk + kh) * Kk + kw]
               : w[((co * CIN_real + ci) * Kk + kh) * Kk + kw];
  }
  out[idx] = (_Float16)v;
}

// =============== tap coordinate helper (conv / transposed conv) ===============
template<int K, int S, int P, bool TRANSP>
__device__ __forceinline__ bool tap_coords(int oh, int ow, int tap, int inH, int inW,
                                           int& ih, int& iw)
{
  if (tap >= K * K) return false;
  const int kh = tap / K, kw = tap % K;
  if constexpr (!TRANSP) {
    ih = oh * S - P + kh;
    iw = ow * S - P + kw;
    return (ih >= 0 && ih < inH && iw >= 0 && iw < inW);
  } else {
    int t1 = oh + P - kh;
    if (t1 < 0 || (t1 % S) != 0) return false;
    ih = t1 / S;
    if (ih >= inH) return false;
    int t2 = ow + P - kw;
    if (t2 < 0 || (t2 % S) != 0) return false;
    iw = t2 / S;
    return iw < inW;
  }
}

// =============== WMMA implicit-GEMM conv: f16 NHWC in -> f16 NHWC out ===============
// One wave = 16 output positions x (COT*16) output channels. Reduction over taps*CI
// in K=32 chunks on v_wmma_f32_16x16x32_f16. Fused bias + BN sum/sumsq reduction.
template<int K, int S, int P, int CI, int COT, bool TRANSP>
__global__ void __launch_bounds__(256)
conv_wmma(const _Float16* __restrict__ inq, const _Float16* __restrict__ wrb,
          const float* __restrict__ bias, _Float16* __restrict__ outq,
          float* __restrict__ st, int inH, int inW, int outH, int outW)
{
  constexpr int TAPS   = K * K;
  constexpr int NCHUNK = (TAPS * CI + 31) / 32;
  constexpr int COUTt  = COT * 16;

  __shared__ _Float16 s_w[NCHUNK * COUTt * 32];
  __shared__ float s_sum[COUTt], s_sq[COUTt];
  const int tid = threadIdx.x;
  if (tid < COUTt) { s_sum[tid] = 0.f; s_sq[tid] = 0.f; }
  {
    const uint4* gw = (const uint4*)wrb;
    uint4* sw = (uint4*)s_w;
    constexpr int TOT = NCHUNK * COUTt * 32 / 8;
    for (int i = tid; i < TOT; i += 256) sw[i] = gw[i];
  }
  __syncthreads();

  const int wid = tid >> 5, lane = tid & 31;
  const int m = lane & 15, khalf = lane >> 4;
  const int outHW = outH * outW;
  const int p0 = (blockIdx.x * 8 + wid) * 16;
  const int p = p0 + m;
  const int n = p / outHW, rem = p % outHW;
  const int oh = rem / outW, ow = rem % outW;

  auto load8 = [&](int tap, int cb) -> uint4 {
    int ih, iw;
    if (tap_coords<K, S, P, TRANSP>(oh, ow, tap, inH, inW, ih, iw))
      return *(const uint4*)(inq + ((size_t)(n * inH + ih) * inW + iw) * CI + cb);
    return make_uint4(0u, 0u, 0u, 0u);
  };
  auto load4 = [&](int tap) -> uint2 {
    int ih, iw;
    if (tap_coords<K, S, P, TRANSP>(oh, ow, tap, inH, inW, ih, iw))
      return *(const uint2*)(inq + ((size_t)(n * inH + ih) * inW + iw) * 4);
    return make_uint2(0u, 0u);
  };

  v8f zeroAcc = {};
  v8f acc[COT];
  #pragma unroll
  for (int c = 0; c < COT; ++c) acc[c] = zeroAcc;

  #pragma unroll
  for (int t = 0; t < NCHUNK; ++t) {
    // A tile per ISA layout: lanes 0..15: local_k {0..7,16..23}; lanes 16..31: {8..15,24..31}
    v16h a;
    if constexpr (CI == 16) {
      uint4* ag = (uint4*)&a;
      ag[0] = load8(2 * t,     8 * khalf);   // group0: tap 2t,   ci = 8*khalf + i
      ag[1] = load8(2 * t + 1, 8 * khalf);   // group1: tap 2t+1, ci = 8*khalf + i
    } else if constexpr (CI == 32) {
      uint4* ag = (uint4*)&a;
      ag[0] = load8(t, 8 * khalf);           // ci = 8*khalf + i
      ag[1] = load8(t, 16 + 8 * khalf);      // ci = 16 + 8*khalf + i
    } else {  // CI == 4
      uint2* ag2 = (uint2*)&a;
      ag2[0] = load4(8 * t + 2 * khalf);
      ag2[1] = load4(8 * t + 2 * khalf + 1);
      ag2[2] = load4(8 * t + 4 + 2 * khalf);
      ag2[3] = load4(8 * t + 4 + 2 * khalf + 1);
    }
    #pragma unroll
    for (int c = 0; c < COT; ++c) {
      // B tile: lanes 0..15 hold K {0..7,16..23}, lanes 16..31 hold K {8..15,24..31} of column co
      const _Float16* bp = s_w + ((size_t)(t * COUTt + (lane & 15) + c * 16) * 32) + khalf * 8;
      v16h b;
      *((uint4*)&b)     = *(const uint4*)bp;
      *((uint4*)&b + 1) = *(const uint4*)(bp + 16);
      acc[c] = __builtin_amdgcn_wmma_f32_16x16x32_f16(false, a, false, b,
                                                      (short)0, acc[c], false, false);
    }
  }

  // bias, f16 rounding, store NHWC, fused BN statistics
  #pragma unroll
  for (int c = 0; c < COT; ++c) {
    const int co = (lane & 15) + c * 16;
    const float bi = bias[co];
    float s = 0.f, q = 0.f;
    _Float16 hv[8];
    #pragma unroll
    for (int r = 0; r < 8; ++r) {
      float v = acc[c][r] + bi;
      _Float16 h = (_Float16)v;
      hv[r] = h;
      float vf = (float)h;
      s += vf; q += vf * vf;
    }
    #pragma unroll
    for (int r = 0; r < 8; ++r) {
      // C layout: lane column N=co, VGPR r -> row M = r + 8*khalf
      outq[(size_t)(p0 + r + 8 * khalf) * COUTt + co] = hv[r];
    }
    s += __shfl_xor(s, 16, 32);  // combine the two lanes holding this channel
    q += __shfl_xor(q, 16, 32);
    if (khalf == 0) { atomicAdd(&s_sum[co], s); atomicAdd(&s_sq[co], q); }
  }
  __syncthreads();
  if (tid < COUTt) {
    atomicAdd(&st[tid], s_sum[tid]);
    atomicAdd(&st[64 + tid], s_sq[tid]);
  }
}

// =============== in-place training-mode BatchNorm + ReLU on f16 NHWC ===============
__global__ void bn_relu_h(_Float16* __restrict__ h, const float* __restrict__ g,
                          const float* __restrict__ be, const float* __restrict__ st,
                          int CH, size_t total, float invCnt)
{
  size_t i = (size_t)blockIdx.x * 256 + threadIdx.x;
  if (i >= total) return;
  int ch = (int)(i % (size_t)CH);
  float mean = st[ch] * invCnt;
  float var  = st[64 + ch] * invCnt - mean * mean;  // biased var, matches jnp.var
  float v = g[ch] * ((float)h[i] - mean) * rsqrtf(var + BN_EPS) + be[ch];
  h[i] = (_Float16)fmaxf(v, 0.f);
}

// e4 variant: also emits the f32 NCHW z_e_x output
__global__ void bn_relu_ze(_Float16* __restrict__ h, float* __restrict__ zef32,
                           const float* __restrict__ g, const float* __restrict__ be,
                           const float* __restrict__ st, float invCnt)
{
  int i = blockIdx.x * 256 + threadIdx.x;
  if (i >= 1183744) return;
  int ch = i & 31;
  float mean = st[ch] * invCnt;
  float var  = st[64 + ch] * invCnt - mean * mean;
  float v = g[ch] * ((float)h[i] - mean) * rsqrtf(var + BN_EPS) + be[ch];
  v = fmaxf(v, 0.f);
  h[i] = (_Float16)v;
  int p = i >> 5, n = p / 289, rem = p % 289;
  zef32[((size_t)(n * 32 + ch)) * 289 + rem] = v;
}

// =============== VQ prep: f16 codebook + squared norms ===============
__global__ void prep_emb(const float* __restrict__ emb, _Float16* __restrict__ ef,
                         float* __restrict__ cn)
{
  int k = blockIdx.x * 256 + threadIdx.x;
  if (k >= 512) return;
  float s = 0.f;
  #pragma unroll
  for (int z = 0; z < 32; ++z) {
    float v = emb[k * 32 + z];
    s += v * v;
    ef[k * 32 + z] = (_Float16)v;
  }
  cn[k] = s;
}

// =============== VQ nearest-codebook search via WMMA f16 ===============
__global__ void __launch_bounds__(256)
vq_argmin_wmma(const _Float16* __restrict__ zef16, const _Float16* __restrict__ embf16,
               const float* __restrict__ cnorm, float* __restrict__ lat_out)
{
  __shared__ _Float16 s_emb[512 * 32];
  __shared__ float    s_cn[512];
  const int tid = threadIdx.x;

  const uint4* ge = (const uint4*)embf16;
  uint4* se = (uint4*)s_emb;
  #pragma unroll
  for (int i = 0; i < 8; ++i) se[tid + i * 256] = ge[tid + i * 256];
  s_cn[tid]       = cnorm[tid];
  s_cn[tid + 256] = cnorm[tid + 256];
  __syncthreads();

  const int wid = tid >> 5, lane = tid & 31;
  const int m = lane & 15;
  const int khalf = lane >> 4;
  const int p0 = (blockIdx.x * 8 + wid) * 16;  // 289 blocks * 8 waves * 16 = 36992

  const _Float16* rowp = zef16 + (size_t)(p0 + m) * 32;
  __builtin_prefetch(rowp + 4096, 0, 0);
  v16h a;
  *((uint4*)&a)     = *(const uint4*)(rowp + khalf * 8);
  *((uint4*)&a + 1) = *(const uint4*)(rowp + 16 + khalf * 8);

  float best[8];
  int   bidx[8];
  #pragma unroll
  for (int r = 0; r < 8; ++r) { best[r] = 3.4e38f; bidx[r] = 0; }

  const int nB = lane & 15;
  const int kb = khalf * 8;

  #pragma unroll 8
  for (int j = 0; j < 32; ++j) {
    const int code = j * 16 + nB;
    const _Float16* bp = s_emb + code * 32 + kb;
    v16h bmat;
    *((uint4*)&bmat)     = *(const uint4*)bp;
    *((uint4*)&bmat + 1) = *(const uint4*)(bp + 16);
    v8f c = {};
    c = __builtin_amdgcn_wmma_f32_16x16x32_f16(false, a, false, bmat,
                                               (short)0, c, false, false);
    const float cn = s_cn[code];
    #pragma unroll
    for (int r = 0; r < 8; ++r) {
      float val = cn - 2.0f * c[r];
      if (val < best[r]) { best[r] = val; bidx[r] = code; }
    }
  }

  #pragma unroll
  for (int r = 0; r < 8; ++r) {
    float v = best[r];
    int   ix = bidx[r];
    #pragma unroll
    for (int mk = 1; mk <= 8; mk <<= 1) {
      float ov = __shfl_xor(v, mk, 32);
      int   oi = __shfl_xor(ix, mk, 32);
      if (ov < v || (ov == v && oi < ix)) { v = ov; ix = oi; }
    }
    if (nB == 0) lat_out[p0 + r + 8 * khalf] = (float)ix;
  }
}

// =============== z_q = emb[latents]: f32 NCHW to d_out + f16 NHWC for decoder ===============
__global__ void gather_zq(const float* __restrict__ emb, const float* __restrict__ lat,
                          float* __restrict__ zq, _Float16* __restrict__ zq16)
{
  int i = blockIdx.x * 256 + threadIdx.x;
  if (i >= 1183744) return;
  int n = i / 9248, r = i % 9248;
  int z = r / 289, rem = r % 289;
  int k = (int)lat[n * 289 + rem];
  float v = emb[k * 32 + z];
  zq[i] = v;
  zq16[(size_t)(n * 289 + rem) * 32 + z] = (_Float16)v;
}

// =============== final 1x1 convT (16 -> 3) + bias (N=3: VALU dot) ===============
__global__ void out_conv(const _Float16* __restrict__ h, const float* __restrict__ w,
                         const float* __restrict__ b, float* __restrict__ y)
{
  int idx = blockIdx.x * 256 + threadIdx.x;
  if (idx >= 6690816) return;
  int n = idx / (3 * 17424);
  int r = idx % (3 * 17424);
  int co = r / 17424, pos = r % 17424;
  const _Float16* hp = h + (size_t)(n * 17424 + pos) * 16;
  float acc = b[co];
  #pragma unroll
  for (int ci = 0; ci < 16; ++ci)
    acc += w[ci * 3 + co] * (float)hp[ci];
  y[idx] = acc;
}

extern "C" void kernel_launch(void* const* d_in, const int* in_sizes, int n_in,
                              void* d_out, int out_size, void* d_ws, size_t ws_size,
                              hipStream_t stream) {
  const float* X    = (const float*)d_in[0];
  const float* E1W  = (const float*)d_in[1],  *E1B = (const float*)d_in[2];
  const float* E1G  = (const float*)d_in[3],  *E1E = (const float*)d_in[4];
  const float* E2W  = (const float*)d_in[5],  *E2B = (const float*)d_in[6];
  const float* E2G  = (const float*)d_in[7],  *E2E = (const float*)d_in[8];
  const float* E3W  = (const float*)d_in[9],  *E3B = (const float*)d_in[10];
  const float* E3G  = (const float*)d_in[11], *E3E = (const float*)d_in[12];
  const float* E4W  = (const float*)d_in[13], *E4B = (const float*)d_in[14];
  const float* E4G  = (const float*)d_in[15], *E4E = (const float*)d_in[16];
  const float* EMB  = (const float*)d_in[17];
  const float* D1W  = (const float*)d_in[18], *D1B = (const float*)d_in[19];
  const float* D1G  = (const float*)d_in[20], *D1E = (const float*)d_in[21];
  const float* D2W  = (const float*)d_in[22], *D2B = (const float*)d_in[23];
  const float* D2G  = (const float*)d_in[24], *D2E = (const float*)d_in[25];
  const float* D3W  = (const float*)d_in[26], *D3B = (const float*)d_in[27];
  const float* D3G  = (const float*)d_in[28], *D3E = (const float*)d_in[29];
  const float* D4W  = (const float*)d_in[30], *D4B = (const float*)d_in[31];
  const float* D4G  = (const float*)d_in[32], *D4E = (const float*)d_in[33];
  const float* OW   = (const float*)d_in[34], *OB  = (const float*)d_in[35];

  _Float16* wsH = (_Float16*)d_ws;
  float*    wsF = (float*)d_ws;
  _Float16* x4  = wsH + H_X4;
  _Float16* a1  = wsH + H_A1;
  _Float16* a2  = wsH + H_A2;
  _Float16* a3  = wsH + H_A3;
  _Float16* ze16 = wsH + H_ZE;
  _Float16* zq16 = wsH + H_ZQ;
  _Float16* b1  = wsH + H_B1;
  _Float16* b2  = wsH + H_B2;
  _Float16* b3  = wsH + H_B3;
  _Float16* b4  = wsH + H_B4;
  _Float16* wr0 = wsH + H_WR + 0 * 4096;
  _Float16* wr1 = wsH + H_WR + 1 * 4096;
  _Float16* wr2 = wsH + H_WR + 2 * 4096;
  _Float16* wr3 = wsH + H_WR + 3 * 4096;
  _Float16* wr4 = wsH + H_WR + 4 * 4096;
  _Float16* wr5 = wsH + H_WR + 5 * 4096;
  _Float16* wr6 = wsH + H_WR + 6 * 4096;
  _Float16* wr7 = wsH + H_WR + 7 * 4096;
  _Float16* embf16 = wsH + H_EMB;
  float* stats = wsF + F_STATS;
  float* cnorm = wsF + F_CNORM;

  float* out   = (float*)d_out;
  float* o_xt  = out + OUT_XT;
  float* o_ze  = out + OUT_ZE;
  float* o_zq  = out + OUT_ZQ;
  float* o_lat = out + OUT_LAT;

  zero_stats<<<4, 256, 0, stream>>>(stats);
  prep_x4<<<32768, 256, 0, stream>>>(X, x4);

  // chunked weights: (w, out, CIN_real, CI, COUT, K, transp, total=NCHUNK*COUT*32)
  prep_wchunks<<<4, 256, 0, stream>>>(E1W, wr0, 3, 4, 16, 4, 0, 1024);
  prep_wchunks<<<16, 256, 0, stream>>>(E2W, wr1, 16, 16, 16, 4, 0, 4096);
  prep_wchunks<<<4, 256, 0, stream>>>(E3W, wr2, 16, 16, 16, 2, 0, 1024);
  prep_wchunks<<<4, 256, 0, stream>>>(E4W, wr3, 16, 16, 32, 1, 0, 1024);
  prep_wchunks<<<2, 256, 0, stream>>>(D1W, wr4, 32, 32, 16, 1, 1, 512);
  prep_wchunks<<<16, 256, 0, stream>>>(D2W, wr5, 16, 16, 16, 4, 1, 4096);
  prep_wchunks<<<4, 256, 0, stream>>>(D3W, wr6, 16, 16, 16, 2, 1, 1024);
  prep_wchunks<<<4, 256, 0, stream>>>(D4W, wr7, 16, 16, 16, 2, 1, 1024);

  // -------- encoder (all WMMA) --------
  conv_wmma<4, 2, 1, 4, 1, false><<<4096, 256, 0, stream>>>(
      x4, wr0, E1B, a1, stats + 0, 128, 128, 64, 64);
  bn_relu_h<<<32768, 256, 0, stream>>>(a1, E1G, E1E, stats + 0, 16,
                                       (size_t)8388608, 1.f / 524288.f);

  conv_wmma<4, 2, 1, 16, 1, false><<<1024, 256, 0, stream>>>(
      a1, wr1, E2B, a2, stats + 128, 64, 64, 32, 32);
  bn_relu_h<<<8192, 256, 0, stream>>>(a2, E2G, E2E, stats + 128, 16,
                                      (size_t)2097152, 1.f / 131072.f);

  conv_wmma<2, 2, 1, 16, 1, false><<<289, 256, 0, stream>>>(
      a2, wr2, E3B, a3, stats + 256, 32, 32, 17, 17);
  bn_relu_h<<<2312, 256, 0, stream>>>(a3, E3G, E3E, stats + 256, 16,
                                      (size_t)591872, 1.f / 36992.f);

  conv_wmma<1, 1, 0, 16, 2, false><<<289, 256, 0, stream>>>(
      a3, wr3, E4B, ze16, stats + 384, 17, 17, 17, 17);
  bn_relu_ze<<<4624, 256, 0, stream>>>(ze16, o_ze, E4G, E4E, stats + 384,
                                       1.f / 36992.f);

  // -------- VQ (WMMA) --------
  prep_emb<<<2, 256, 0, stream>>>(EMB, embf16, cnorm);
  vq_argmin_wmma<<<289, 256, 0, stream>>>(ze16, embf16, cnorm, o_lat);
  gather_zq<<<4624, 256, 0, stream>>>(EMB, o_lat, o_zq, zq16);

  // -------- decoder (all WMMA) --------
  conv_wmma<1, 1, 0, 32, 1, true><<<289, 256, 0, stream>>>(
      zq16, wr4, D1B, b1, stats + 512, 17, 17, 17, 17);
  bn_relu_h<<<2312, 256, 0, stream>>>(b1, D1G, D1E, stats + 512, 16,
                                      (size_t)591872, 1.f / 36992.f);

  conv_wmma<4, 2, 1, 16, 1, true><<<1156, 256, 0, stream>>>(
      b1, wr5, D2B, b2, stats + 640, 17, 17, 34, 34);
  bn_relu_h<<<9248, 256, 0, stream>>>(b2, D2G, D2E, stats + 640, 16,
                                      (size_t)2367488, 1.f / 147968.f);

  conv_wmma<2, 2, 1, 16, 1, true><<<4356, 256, 0, stream>>>(
      b2, wr6, D3B, b3, stats + 768, 34, 34, 66, 66);
  bn_relu_h<<<34848, 256, 0, stream>>>(b3, D3G, D3E, stats + 768, 16,
                                       (size_t)8921088, 1.f / 557568.f);

  conv_wmma<2, 2, 0, 16, 1, true><<<17424, 256, 0, stream>>>(
      b3, wr7, D4B, b4, stats + 896, 66, 66, 132, 132);
  bn_relu_h<<<139392, 256, 0, stream>>>(b4, D4G, D4E, stats + 896, 16,
                                        (size_t)35684352, 1.f / 2230272.f);

  out_conv<<<26136, 256, 0, stream>>>(b4, OW, OB, o_xt);
}
